// AttentionBlock_81819126989554
// MI455X (gfx1250) — matrix operand order repaired and checked by reading, not compiled, not run on hardware
//
#include <hip/hip_runtime.h>

typedef _Float16 v16h __attribute__((ext_vector_type(16)));
typedef _Float16 v8h  __attribute__((ext_vector_type(8)));
typedef float    v8f  __attribute__((ext_vector_type(8)));
typedef int      v4i_b __attribute__((vector_size(16)));   // gcc-style, matches builtin

#define T_SEQ 2048
#define HID   2048
#define NHEAD 32
#define NKVH  8
#define HDIM  64
#define SWIN  128
#define QKVD  3072
#define RMS_EPS 1e-5f

#define GLOBAL_AS __attribute__((address_space(1)))
#define LDS_AS    __attribute__((address_space(3)))

// 16-byte global->LDS copy: async (ASYNCcnt) when the gfx1250 builtin exists,
// else plain load + ds_store_b128 (tracked by DScnt / __syncthreads).
__device__ __forceinline__ void copy16_g2l(const _Float16* g, _Float16* l) {
#if __has_builtin(__builtin_amdgcn_global_load_async_to_lds_b128)
  __builtin_amdgcn_global_load_async_to_lds_b128(
      (GLOBAL_AS v4i_b*)g, (LDS_AS v4i_b*)l, 0, 0);
#else
  *(v8h*)l = *(const v8h*)g;
#endif
}

__device__ __forceinline__ void wait_async_copies() {
#if __has_builtin(__builtin_amdgcn_global_load_async_to_lds_b128)
#if __has_builtin(__builtin_amdgcn_s_wait_asynccnt)
  __builtin_amdgcn_s_wait_asynccnt(0);
#else
  asm volatile("s_wait_asynccnt 0x0" ::: "memory");
#endif
#endif
}

// ---------------- RMSNorm + f16 convert (one block per row) ----------------
__global__ __launch_bounds__(256)
void rmsnorm_kernel(const float* __restrict__ x, const float* __restrict__ scale,
                    _Float16* __restrict__ out) {
  int row = blockIdx.x;
  int tid = threadIdx.x;
  const float* xr = x + (size_t)row * HID;
  float vals[8];
  float p = 0.f;
#pragma unroll
  for (int i = 0; i < 8; ++i) {
    float v = xr[tid + i * 256];
    vals[i] = v;
    p += v * v;
  }
#pragma unroll
  for (int o = 16; o > 0; o >>= 1) p += __shfl_xor(p, o, 32);
  __shared__ float red[8];
  int wid = tid >> 5, lane = tid & 31;
  if (lane == 0) red[wid] = p;
  __syncthreads();
  float tot = 0.f;
#pragma unroll
  for (int i = 0; i < 8; ++i) tot += red[i];
  float inv = rsqrtf(tot * (1.0f / (float)HID) + RMS_EPS);
  _Float16* orow = out + (size_t)row * HID;
#pragma unroll
  for (int i = 0; i < 8; ++i) {
    int c = tid + i * 256;
    orow[c] = (_Float16)(vals[i] * inv * scale[c]);
  }
}

// ------- Weight fp32 [K,N] row-major -> f16 WMMA-B fragment-swizzled -------
// Fragment(kb,nb) = 32(K) x 16(N). Lane L: nn=L&15, kk = ((L>>4)<<4)+h, h=0..15.
// Stored contiguously: frag*512 + lane*16 + h halves (32B per lane, coalesced).
__global__ __launch_bounds__(256)
void swizzle_w_kernel(const float* __restrict__ W, _Float16* __restrict__ out,
                      int K, int N) {
  int gid = blockIdx.x * 256 + threadIdx.x;
  int lane = gid & 31;
  int frag = gid >> 5;
  int kfrags = K >> 5;
  int nb = frag / kfrags;
  int kb = frag - nb * kfrags;
  int kbase = kb * 32 + ((lane >> 4) << 4);
  const float* src = W + (size_t)kbase * N + nb * 16 + (lane & 15);
  union { v16h v; v8h h[2]; } f;
#pragma unroll
  for (int h = 0; h < 16; ++h) f.v[h] = (_Float16)src[(size_t)h * N];
  v8h* dst = (v8h*)(out + (size_t)frag * 512 + lane * 16);
  dst[0] = f.h[0];
  dst[1] = f.h[1];
}

// ------ f16 GEMM, LDS double-buffered with async copies: C = A*Bsw + bias --
// Block: 256 thr / 8 waves; tile M=128 N=128, k-step 32.
// Waves: wm = w&1 (2 along M, 64 rows each), wn = w>>1 (4 along N, 32 cols).
#define APITCH 40   // padded halves per A row in LDS (80B -> conflict-free)

template <bool ADD_RES>
__global__ __launch_bounds__(256)
void gemm_f16_kernel(const _Float16* __restrict__ A,
                     const _Float16* __restrict__ Bsw,
                     const float* __restrict__ bias,
                     const float* __restrict__ res,
                     float* __restrict__ C,
                     int N, int K) {
  __shared__ _Float16 As[2][128 * APITCH];   // 2 x 10 KB
  __shared__ _Float16 Bs[2][128 * 32];       // 2 x 8 KB (8 frags x 512 halves)

  int tid  = threadIdx.x;
  int w    = tid >> 5;
  int lane = tid & 31;
  int wm = w & 1;
  int wn = w >> 1;
  int rowBase = blockIdx.x * 128;
  int colBase = blockIdx.y * 128;
  int kfrags  = K >> 5;
  int kiters  = K >> 5;

  // ---- per-thread copy assignments (whole block cooperates) ----
  // A tile: 128 rows x 32 halves; thread t: row=t>>1, halves [(t&1)*16, +16)
  int arow = tid >> 1;
  int acol = (tid & 1) * 16;
  const _Float16* gA = A + (size_t)(rowBase + arow) * K + acol;
  // B tile: 8 frags x 1KB, already fragment-contiguous in Bsw
  int bfrag = tid >> 5;
  const _Float16* gB = Bsw + (size_t)((colBase >> 4) + bfrag) * kfrags * 512
                           + lane * 16;

  v8f zero = {};
  v8f acc[4][2];
#pragma unroll
  for (int i = 0; i < 4; ++i)
#pragma unroll
    for (int j = 0; j < 2; ++j) acc[i][j] = zero;

  int r    = lane & 15;
  int ksel = (lane >> 4) << 4;   // lanes 16..31 take the K+16 halves (contiguous layout)

  // prologue: stage k-tile 0 into buffer 0
  {
    _Float16* la = &As[0][arow * APITCH + acol];
    copy16_g2l(gA, la);
    copy16_g2l(gA + 8, la + 8);
    _Float16* lb = &Bs[0][bfrag * 512 + lane * 16];
    copy16_g2l(gB, lb);
    copy16_g2l(gB + 8, lb + 8);
  }

  int buf = 0;
  for (int kb = 0; kb < kiters; ++kb) {
    wait_async_copies();
    __syncthreads();                 // buffer `buf` ready; previous reads done

    if (kb + 1 < kiters) {           // kick next tile into the other buffer
      const _Float16* ga = gA + ((kb + 1) << 5);
      _Float16* la = &As[buf ^ 1][arow * APITCH + acol];
      copy16_g2l(ga, la);
      copy16_g2l(ga + 8, la + 8);
      const _Float16* gb = gB + (size_t)(kb + 1) * 512;
      _Float16* lb = &Bs[buf ^ 1][bfrag * 512 + lane * 16];
      copy16_g2l(gb, lb);
      copy16_g2l(gb + 8, lb + 8);
    }

    union { v16h v; v8h h[2]; } a[4];
#pragma unroll
    for (int mi = 0; mi < 4; ++mi) {
      const _Float16* pa = &As[buf][(wm * 64 + mi * 16 + r) * APITCH + ksel];
      a[mi].h[0] = *(const v8h*)pa;           // ds_load_b128
      a[mi].h[1] = *(const v8h*)(pa + 8);
    }
    union { v16h v; v8h h[2]; } b[2];
#pragma unroll
    for (int ni = 0; ni < 2; ++ni) {
      const _Float16* pb = &Bs[buf][(wn * 2 + ni) * 512 + lane * 16];
      b[ni].h[0] = *(const v8h*)pb;
      b[ni].h[1] = *(const v8h*)(pb + 8);
    }
#pragma unroll
    for (int mi = 0; mi < 4; ++mi)
#pragma unroll
      for (int ni = 0; ni < 2; ++ni)
        acc[mi][ni] = __builtin_amdgcn_wmma_f32_16x16x32_f16(
            false, a[mi].v, false, b[ni].v, (short)0, acc[mi][ni], false, false);

    buf ^= 1;
  }

  // C/D layout: VGPR e -> M = e (+8 for lanes>=16); N = lane&15
  int cn   = lane & 15;
  int rsel = (lane >> 4) << 3;
#pragma unroll
  for (int mi = 0; mi < 4; ++mi) {
#pragma unroll
    for (int ni = 0; ni < 2; ++ni) {
      int col = colBase + wn * 32 + ni * 16 + cn;
      float bcol = bias[col];
#pragma unroll
      for (int e = 0; e < 8; ++e) {
        int row = rowBase + wm * 64 + mi * 16 + rsel + e;
        size_t idx = (size_t)row * N + col;
        float v = acc[mi][ni][e] + bcol;
        if (ADD_RES) v += res[idx];
        C[idx] = v;
      }
    }
  }
}

// ---------------- RoPE on q (heads 0..31) and k (heads 32..39) -------------
__global__ __launch_bounds__(256)
void rope_kernel(float* __restrict__ qkv, const float* __restrict__ cosT,
                 const float* __restrict__ sinT) {
  int t = blockIdx.x;
  for (int it = threadIdx.x; it < 40 * 32; it += 256) {
    int hh = it >> 5;
    int d  = it & 31;
    int off = (hh < NHEAD) ? hh * HDIM : (NHEAD * HDIM + (hh - NHEAD) * HDIM);
    float c = cosT[t * 32 + d];
    float s = sinT[t * 32 + d];
    float* p = qkv + (size_t)t * QKVD + off;
    float x1 = p[d], x2 = p[d + 32];
    p[d]      = x1 * c - x2 * s;
    p[d + 32] = x2 * c + x1 * s;
  }
}

// -------- Sliding-window attention with sink folded into (m,l) state -------
__global__ __launch_bounds__(256)
void attn_kernel(const float* __restrict__ qkv, const float* __restrict__ sinks,
                 _Float16* __restrict__ attn) {
  int h = blockIdx.y;
  int t = blockIdx.x * 256 + threadIdx.x;
  int kv = h >> 2;  // QM = 4
  const float* qp = qkv + (size_t)t * QKVD + h * HDIM;
  float q[HDIM], acc[HDIM];
#pragma unroll
  for (int d = 0; d < HDIM; ++d) { q[d] = qp[d]; acc[d] = 0.f; }
  float m = sinks[h];   // sink logit seeds the running max
  float l = 1.0f;       // exp(sink - m) = 1
  int j0 = t - SWIN; if (j0 < 0) j0 = 0;
  for (int j = j0; j <= t; ++j) {
    const float* kp = qkv + (size_t)j * QKVD + (NHEAD * HDIM) + kv * HDIM;
    float dot = 0.f;
#pragma unroll
    for (int d = 0; d < HDIM; ++d) dot += q[d] * kp[d];
    dot *= 0.125f;  // 1/sqrt(64)
    float mn = fmaxf(m, dot);
    float corr = __expf(m - mn);
    float e    = __expf(dot - mn);
    l = l * corr + e;
    m = mn;
    const float* vp = qkv + (size_t)j * QKVD + (NHEAD + NKVH) * HDIM + kv * HDIM;
#pragma unroll
    for (int d = 0; d < HDIM; ++d) acc[d] = acc[d] * corr + e * vp[d];
  }
  float invl = 1.0f / l;
  _Float16* op = attn + (size_t)t * HID + h * HDIM;
#pragma unroll
  for (int d = 0; d < HDIM; ++d) op[d] = (_Float16)(acc[d] * invl);
}

// ---------------------------------------------------------------------------
extern "C" void kernel_launch(void* const* d_in, const int* in_sizes, int n_in,
                              void* d_out, int out_size, void* d_ws, size_t ws_size,
                              hipStream_t stream) {
  (void)in_sizes; (void)n_in; (void)out_size; (void)ws_size;
  const float* x      = (const float*)d_in[0];
  const float* scale  = (const float*)d_in[1];
  const float* sinks  = (const float*)d_in[2];
  const float* wqkv   = (const float*)d_in[3];
  const float* bqkv   = (const float*)d_in[4];
  const float* wout   = (const float*)d_in[5];
  const float* bout   = (const float*)d_in[6];
  const float* cosT   = (const float*)d_in[7];
  const float* sinT   = (const float*)d_in[8];
  // d_in[9] (mask) handled analytically by the windowed attention loop.

  char* wsb = (char*)d_ws;
  _Float16* normed = (_Float16*)(wsb);                         //  8 MB
  _Float16* wqkv_h = (_Float16*)(wsb + ((size_t)8  << 20));    // 12 MB
  _Float16* wout_h = (_Float16*)(wsb + ((size_t)20 << 20));    //  8 MB
  float*    qkvbuf = (float*)   (wsb + ((size_t)28 << 20));    // 24 MB
  _Float16* attnb  = (_Float16*)(wsb + ((size_t)52 << 20));    //  8 MB

  rmsnorm_kernel<<<T_SEQ, 256, 0, stream>>>(x, scale, normed);

  // qkv weights: frags = (2048/32)*(3072/16) = 12288 -> 1536 blocks
  swizzle_w_kernel<<<1536, 256, 0, stream>>>(wqkv, wqkv_h, HID, QKVD);
  // out weights: frags = 64*128 = 8192 -> 1024 blocks
  swizzle_w_kernel<<<1024, 256, 0, stream>>>(wout, wout_h, HID, HID);

  dim3 g1(T_SEQ / 128, QKVD / 128);
  gemm_f16_kernel<false><<<g1, 256, 0, stream>>>(normed, wqkv_h, bqkv, nullptr,
                                                 qkvbuf, QKVD, HID);

  rope_kernel<<<T_SEQ, 256, 0, stream>>>(qkvbuf, cosT, sinT);

  dim3 ga(T_SEQ / 256, NHEAD);
  attn_kernel<<<ga, 256, 0, stream>>>(qkvbuf, sinks, attnb);

  dim3 g2(T_SEQ / 128, HID / 128);
  gemm_f16_kernel<true><<<g2, 256, 0, stream>>>(attnb, wout_h, bout, x,
                                                (float*)d_out, HID, HID);
}
